// PureTriadicBrain_73753178407495
// MI455X (gfx1250) — compile-verified, matrix operand order checked by reading, not run on hardware
//
#include <hip/hip_runtime.h>
#include <hip/hip_bf16.h>
#include <math.h>

// ---------------------------------------------------------------------------
// PureTriadicBrain for MI455X (gfx1250, wave32).
// Memory-bound: ~1.15 GB weight traffic dominates (~50us floor @ 23.3 TB/s).
// GEMMs (QK^T, A@O) run on v_wmma_f32_16x16x32_f16 with all-contiguous
// v16h (32B) fragment loads; A@O uses a pre-transposed O operand.
// Weight update fuses rank-1 gradient + WD + mask + momentum + rms_norm,
// stashing W_tmp as f16 in LDS (32KB/block) to avoid a second global pass.
// Sinkhorn column pass parallelized via per-column atomics (the naive
// column-owner version would serialize on ~2K threads).
// ---------------------------------------------------------------------------

#define B_ 4
#define N_ 512
#define D_ 128
static const int    SND = B_ * N_ * D_;                    // 262144
static const int    SNN = B_ * N_ * N_;                    // 1048576
static const long long SWD = (long long)B_ * N_ * D_ * D_; // 33554432

#define UNTANH_ 3.0f
#define BUNT_   1.84775906502257351226f   // sqrt(2 + sqrt(2))
#define INV_SQRT_D_ 0.08838834764831845f  // 1/sqrt(128)

typedef _Float16 v16h __attribute__((ext_vector_type(16)));
typedef float    v8f  __attribute__((ext_vector_type(8)));

// --------------------------- hash RNG (stand-in) ---------------------------
__device__ __forceinline__ unsigned hash_u32(unsigned x) {
  x ^= x >> 16; x *= 0x7feb352dU; x ^= x >> 15; x *= 0x846ca68bU; x ^= x >> 16;
  return x;
}
__device__ __forceinline__ float randn_hash(unsigned idx, unsigned seed) {
  unsigned h1 = hash_u32(idx * 0x9E3779B1U + seed);
  unsigned h2 = hash_u32(h1 ^ 0x85EBCA6BU);
  float u1 = (float)(h1 & 0x00FFFFFFu) * (1.0f / 16777216.0f) + 1e-7f;
  float u2 = (float)(h2 & 0x00FFFFFFu) * (1.0f / 16777216.0f);
  return sqrtf(-2.0f * logf(u1)) * __cosf(6.2831853f * u2);
}

// --------------------- 1) piece_forward (3 pieces fused) -------------------
// out[b,n,k] = buntanh( sum_j state[b,n,j] * W[b,n,j,k] )
__global__ void fwd_kernel(const float* __restrict__ state,
                           const float* __restrict__ w1,
                           const float* __restrict__ w2,
                           const float* __restrict__ w3,
                           float* __restrict__ pred,
                           float* __restrict__ keys,
                           float* __restrict__ queries) {
  int bn = blockIdx.x, t = threadIdx.x, p = blockIdx.y;
  const float* W = (p == 0) ? w1 : (p == 1) ? w2 : w3;
  float* out = (p == 0) ? pred : (p == 1) ? keys : queries;
  __shared__ float sx[D_];
  sx[t] = state[(size_t)bn * D_ + t];
  __syncthreads();
  const float* Wb = W + (size_t)bn * D_ * D_;
  float acc = 0.f;
#pragma unroll 4
  for (int j = 0; j < D_; ++j) acc = fmaf(sx[j], Wb[(size_t)j * D_ + t], acc);
  out[(size_t)bn * D_ + t] = UNTANH_ * tanhf(acc * (1.0f / UNTANH_)) * BUNT_;
}

// --------------------------- f32 -> f16 converts ---------------------------
__global__ void tohalf_qk_kernel(const float* __restrict__ q, const float* __restrict__ k,
                                 _Float16* __restrict__ qh, _Float16* __restrict__ kh) {
  int i = blockIdx.x * 256 + threadIdx.x;
  qh[i] = (_Float16)q[i];
  kh[i] = (_Float16)k[i];
}
__global__ void convA_kernel(const float* __restrict__ A, _Float16* __restrict__ Ah) {
  int i = blockIdx.x * 256 + threadIdx.x;
  Ah[i] = (_Float16)A[i];
}
// (output + boredom_noise) -> f16, TRANSPOSED to [B, D, N] so the WMMA
// B-fragment for target = A@O becomes a contiguous 32B load per lane.
__global__ void convO_kernel(const float* __restrict__ outp,
                             const float* __restrict__ vsum, const float* __restrict__ vsq,
                             _Float16* __restrict__ OhT) {
  int i = blockIdx.x * 256 + threadIdx.x;   // index in transposed [B,D,N] layout
  int b = i / (D_ * N_);
  int r = i - b * (D_ * N_);
  int d = r / N_;
  int n = r - d * N_;
  float inv = 1.0f / (float)(N_ * N_);
  float mean = vsum[b] * inv;
  float var = vsq[b] * inv - mean * mean;
  float vd = fmaxf(0.01f - var, 0.0f);                  // relu(A_VAR_GOAL - var)
  float v = outp[((size_t)b * N_ + n) * D_ + d];
  OhT[i] = (_Float16)(v + vd * randn_hash((unsigned)i, 0xB0DEC0DEu));
}

// ------------------- 2) raw_A = Q K^T / sqrt(D)  (WMMA) --------------------
// One wave per 16x16 tile. Lane l: row = tile + (l&15); K-halves contiguous.
__global__ void wmma_qk_kernel(const _Float16* __restrict__ qh,
                               const _Float16* __restrict__ kh,
                               float* __restrict__ rawA) {
  int b = blockIdx.z, m0 = blockIdx.y * 16, n0 = blockIdx.x * 16;
  int lane = threadIdx.x, lo = lane & 15, klo = (lane < 16) ? 0 : 16;
  const _Float16* arow = qh + ((size_t)b * N_ + (m0 + lo)) * D_;
  const _Float16* brow = kh + ((size_t)b * N_ + (n0 + lo)) * D_;
  v8f c = {};
#pragma unroll
  for (int kk = 0; kk < 4; ++kk) {                       // K = 128 in steps of 32
    int kb = kk * 32 + klo;
    v16h a = *(const v16h*)(arow + kb);                  // 32B aligned
    v16h bf = *(const v16h*)(brow + kb);
    c = __builtin_amdgcn_wmma_f32_16x16x32_f16(false, a, false, bf,
                                               (short)0, c, false, false);
  }
#pragma unroll
  for (int r = 0; r < 8; ++r) {
    int row = m0 + r + ((lane >= 16) ? 8 : 0);
    rawA[((size_t)b * N_ + row) * N_ + n0 + lo] = c[r] * INV_SQRT_D_;
  }
}

// -------------- 3) target = A @ (output+noise)  (WMMA, O^T) ----------------
__global__ void wmma_tgt_kernel(const _Float16* __restrict__ Ah,
                                const _Float16* __restrict__ OhT,
                                float* __restrict__ tgt) {
  int b = blockIdx.z, m0 = blockIdx.y * 16, n0 = blockIdx.x * 16;
  int lane = threadIdx.x, lo = lane & 15, klo = (lane < 16) ? 0 : 16;
  const _Float16* arow = Ah  + ((size_t)b * N_ + (m0 + lo)) * N_;
  const _Float16* brow = OhT + ((size_t)b * D_ + (n0 + lo)) * N_;  // col-contig K
  v8f c = {};
#pragma unroll
  for (int kk = 0; kk < 16; ++kk) {                      // K = 512 in steps of 32
    int kb = kk * 32 + klo;
    v16h a  = *(const v16h*)(arow + kb);
    v16h bf = *(const v16h*)(brow + kb);
    c = __builtin_amdgcn_wmma_f32_16x16x32_f16(false, a, false, bf,
                                               (short)0, c, false, false);
  }
#pragma unroll
  for (int r = 0; r < 8; ++r) {
    int row = m0 + r + ((lane >= 16) ? 8 : 0);
    tgt[((size_t)b * N_ + row) * D_ + n0 + lo] = c[r];
  }
}

// ----------------------- per-b reductions (atomics) ------------------------
__global__ void ssq_kernel(const float* __restrict__ X, float* __restrict__ acc) {
  int b = blockIdx.y, t = threadIdx.x;
  size_t base = (size_t)b * N_ * N_ + (size_t)blockIdx.x * 4096;
  float s = 0.f;
#pragma unroll 4
  for (int i = 0; i < 16; ++i) { float v = X[base + i * 256 + t]; s += v * v; }
  __shared__ float red[256];
  red[t] = s; __syncthreads();
  for (int k = 128; k > 0; k >>= 1) { if (t < k) red[t] += red[t + k]; __syncthreads(); }
  if (t == 0) atomicAdd(acc + b, red[0]);
}
__global__ void vstat_kernel(const float* __restrict__ X,
                             float* __restrict__ vsum, float* __restrict__ vsq) {
  int b = blockIdx.y, t = threadIdx.x;
  size_t base = (size_t)b * N_ * N_ + (size_t)blockIdx.x * 4096;
  float s = 0.f, q = 0.f;
#pragma unroll 4
  for (int i = 0; i < 16; ++i) { float v = X[base + i * 256 + t]; s += v; q += v * v; }
  __shared__ float r1[256], r2[256];
  r1[t] = s; r2[t] = q; __syncthreads();
  for (int k = 128; k > 0; k >>= 1) {
    if (t < k) { r1[t] += r1[t + k]; r2[t] += r2[t + k]; }
    __syncthreads();
  }
  if (t == 0) { atomicAdd(vsum + b, r1[0]); atomicAdd(vsq + b, r2[0]); }
}

// ------- rms_norm(raw_A over (-1,-2)) + EMA -> A_ema_new (out + work) ------
__global__ void ema_kernel(const float* __restrict__ rawA, const float* __restrict__ ssq,
                           const float* __restrict__ Aold, const int* __restrict__ step,
                           float* __restrict__ Aout, float* __restrict__ Awork) {
  size_t i = (size_t)blockIdx.x * 256 + threadIdx.x;
  int b = (int)(i / (N_ * N_));
  float sc = rsqrtf(ssq[b] * (1.0f / (float)(N_ * N_)) + 1e-8f);
  float rn = rawA[i] * sc;
  float v = (*step == 1) ? rn : 0.9f * Aold[i] + 0.1f * rn;  // C_MOMENT = 0.9
  Aout[i] = v;
  Awork[i] = v;
}

// ------------------------------- Sinkhorn ----------------------------------
__global__ void sink_row_kernel(float* __restrict__ A) {
  int r = blockIdx.x, t = threadIdx.x;               // r over B*N rows
  size_t base = (size_t)r * N_;
  float v[4], s = 0.f;
#pragma unroll
  for (int c = 0; c < 4; ++c) { v[c] = A[base + c * 128 + t]; s += v[c]; }
  __shared__ float red[128];
  red[t] = s; __syncthreads();
  for (int k = 64; k > 0; k >>= 1) { if (t < k) red[t] += red[t + k]; __syncthreads(); }
  float inv = 1.0f / (red[0] + 1e-6f);
#pragma unroll
  for (int c = 0; c < 4; ++c) A[base + c * 128 + t] = v[c] * inv;
}
// Column pass, phase 1: 8-row strips accumulate per-column partial sums.
__global__ void colsum_kernel(const float* __restrict__ A, float* __restrict__ csum) {
  int b = blockIdx.y, rg = blockIdx.x, t = threadIdx.x;   // rg: 64 strips of 8 rows
  size_t base = (size_t)b * N_ * N_ + (size_t)rg * 8 * N_;
  float s0 = 0.f, s1 = 0.f;
#pragma unroll
  for (int r = 0; r < 8; ++r) {
    s0 += A[base + r * N_ + t];
    s1 += A[base + r * N_ + t + 256];
  }
  atomicAdd(&csum[b * N_ + t], s0);
  atomicAdd(&csum[b * N_ + t + 256], s1);
}
// Column pass, phase 2: fully parallel divide.
__global__ void coldiv_kernel(float* __restrict__ A, const float* __restrict__ csum) {
  int i = blockIdx.x * 256 + threadIdx.x;
  int b = i / (N_ * N_);
  int col = i & (N_ - 1);
  A[i] /= (csum[b * N_ + col] + 1e-6f);
}
__global__ void thresh_kernel(float* __restrict__ A) {
  int i = blockIdx.x * 256 + threadIdx.x;
  float v = A[i];
  A[i] = (v > (0.5f / (float)N_)) ? v : 0.0f;
}
__global__ void zero_rows_kernel(float* __restrict__ A) {
  int b = blockIdx.x, r = blockIdx.y, t = threadIdx.x;   // rows 0,1
  A[((size_t)b * N_ + r) * N_ + t] = 0.0f;
}

// ----------------------------- target env fix ------------------------------
__global__ void envfix_kernel(float* __restrict__ ns,
                              const float* __restrict__ eye,
                              const float* __restrict__ st) {
  int b = blockIdx.x, t = threadIdx.x;
  ns[((size_t)b * N_ + 0) * D_ + t] = eye[b * D_ + t];
  ns[((size_t)b * N_ + 1) * D_ + t] = st[b * D_ + t];
}

// ------------------------------- errors ------------------------------------
__global__ void err_kernel(const float* __restrict__ pred, const float* __restrict__ tgt,
                           const float* __restrict__ keys, const float* __restrict__ qs,
                           float* __restrict__ e1o, float* __restrict__ e2o,
                           float* __restrict__ e3o) {
  int i = blockIdx.x * 256 + threadIdx.x;
  float p = pred[i], tg = tgt[i];
  float e1b = p - tg;
  float e2 = keys[i] - e1b;
  float e3 = qs[i] - e2;
  e1o[i] = p - (tg + e3);
  e2o[i] = e2;
  e3o[i] = e3;
}

// -------- coef_i = -plasticity_i * err_i / sqrt(err_i^2*msq + eps) ---------
// (rms_norm of outer product -err_i*noisy_j over j factorizes: grad = coef_i*noisy_j)
__global__ void coef_kernel(const float* __restrict__ state,
                            const float* __restrict__ err1, const float* __restrict__ err2,
                            const float* __restrict__ err3,
                            const float* __restrict__ eb1, const float* __restrict__ eb2,
                            const float* __restrict__ eb3,
                            float* __restrict__ n1, float* __restrict__ n2, float* __restrict__ n3,
                            float* __restrict__ c1, float* __restrict__ c2, float* __restrict__ c3) {
  int bn = blockIdx.x, t = threadIdx.x, p = blockIdx.y;
  const float* err = (p == 0) ? err1 : (p == 1) ? err2 : err3;
  const float* eb  = (p == 0) ? eb1  : (p == 1) ? eb2  : eb3;
  float* noisy = (p == 0) ? n1 : (p == 1) ? n2 : n3;
  float* coef  = (p == 0) ? c1 : (p == 1) ? c2 : c3;
  size_t idx = (size_t)bn * D_ + t;
  float e = err[idx];
  __shared__ float red[128];
  red[t] = e; __syncthreads();
  for (int k = 64; k > 0; k >>= 1) { if (t < k) red[t] = fmaxf(red[t], red[t + k]); __syncthreads(); }
  float mx = red[0]; __syncthreads();
  float ex = expf(e - mx);
  red[t] = ex; __syncthreads();
  for (int k = 64; k > 0; k >>= 1) { if (t < k) red[t] += red[t + k]; __syncthreads(); }
  float ssum = red[0]; __syncthreads();
  float Ec = ex / ssum;                         // softmax
  float En = eb[idx] * 0.99f + 0.01f * Ec;      // EMA_SPEED = 0.01
  float adv = Ec - En;
  float x = 4.0f * adv;                         // REW_SENS
  float plast = 1.0f + x * rsqrtf(x * x + 1e-8f);
  float nz = state[idx] + 0.01f * randn_hash((unsigned)idx, 0x517CC1B7u * (p + 1));
  noisy[idx] = nz;
  red[t] = nz * nz; __syncthreads();
  for (int k = 64; k > 0; k >>= 1) { if (t < k) red[t] += red[t + k]; __syncthreads(); }
  float msq = red[0] * (1.0f / (float)D_); __syncthreads();
  coef[idx] = -plast * e * rsqrtf(e * e * msq + 1e-8f);
}

// ---- fused piece_step: grad=coef_i*noisy_j, WD, mask, momentum, rms_norm --
// W_tmp stashed as f16 in LDS (32 KB) -> single global read of W,g and single
// write of W_new. Streaming (6x128MB >> 192MB L2) -> software prefetch.
__global__ void __launch_bounds__(256) wstep_kernel(const float* __restrict__ Wp,
                                                    const float* __restrict__ Gp,
                                                    const float* __restrict__ coef,
                                                    const float* __restrict__ noisy,
                                                    const int* __restrict__ step,
                                                    float* __restrict__ out) {
  int bn = blockIdx.x, t = threadIdx.x;
  __shared__ _Float16 wt[D_ * D_];
  __shared__ float sc[D_], sn[D_], sm[D_], red[256];
  if (t < D_) {
    sc[t] = coef[(size_t)bn * D_ + t];
    sn[t] = noisy[(size_t)bn * D_ + t];
    float center = fmodf((float)(*step), (float)D_);   // SPEED = 1.0
    float d = fabsf((float)t - center);
    d = fminf(d, (float)D_ - d);                       // wrap-around distance
    sm[t] = __expf(-d * d * (1.0f / 32.0f));           // 2*WIDTH^2 = 32
  }
  __syncthreads();
  size_t base = (size_t)bn * (D_ * D_);
  const float4* W4 = (const float4*)(Wp + base);
  const float4* G4 = (const float4*)(Gp + base);
  float ss = 0.f;
#pragma unroll 4
  for (int s = 0; s < 16; ++s) {
    int v = s * 256 + t;
    if (s < 12) {                                      // stream prefetch, +16KB
      __builtin_prefetch((const void*)(W4 + (s + 4) * 256 + t), 0, 0);
      __builtin_prefetch((const void*)(G4 + (s + 4) * 256 + t), 0, 0);
    }
    float4 w = W4[v], g = G4[v];
    int off = v * 4, i = off >> 7, j = off & 127;
    float ci = sc[i];
    const float* wp = (const float*)&w;
    const float* gp = (const float*)&g;
#pragma unroll
    for (int c = 0; c < 4; ++c) {
      float grad = (ci * sn[j + c] - 1e-4f * wp[c]) * sm[j + c];  // -WD*W then mask
      float gn = 0.9f * gp[c] + 0.1f * grad;                      // MOMENTUM
      float wnew = wp[c] + 0.01f * gn;                            // LR
      wt[off + c] = (_Float16)wnew;
      ss += wnew * wnew;
    }
  }
  red[t] = ss; __syncthreads();
  for (int k = 128; k > 0; k >>= 1) { if (t < k) red[t] += red[t + k]; __syncthreads(); }
  float scale = 4.0f * rsqrtf(red[0] * (1.0f / (float)(D_ * D_)) + 1e-8f); // W_SCALE
  float4* O4 = (float4*)(out + base);
#pragma unroll 4
  for (int s = 0; s < 16; ++s) {
    int v = s * 256 + t, off = v * 4;
    float4 o;
    float* op = (float*)&o;
#pragma unroll
    for (int c = 0; c < 4; ++c) op[c] = (float)wt[off + c] * scale;
    O4[v] = o;
  }
}

// ===========================================================================
extern "C" void kernel_launch(void* const* d_in, const int* in_sizes, int n_in,
                              void* d_out, int out_size, void* d_ws, size_t ws_size,
                              hipStream_t stream) {
  const float* state   = (const float*)d_in[0];
  const float* outp    = (const float*)d_in[1];
  const float* A_ema   = (const float*)d_in[2];
  const float* w1      = (const float*)d_in[3];
  const float* w2      = (const float*)d_in[4];
  const float* w3      = (const float*)d_in[5];
  const float* g1      = (const float*)d_in[6];
  const float* g2      = (const float*)d_in[7];
  const float* g3      = (const float*)d_in[8];
  const float* eb1     = (const float*)d_in[9];
  const float* eb2     = (const float*)d_in[10];
  const float* eb3     = (const float*)d_in[11];
  const float* eye     = (const float*)d_in[12];
  const float* stomach = (const float*)d_in[13];
  const int*   step    = (const int*)d_in[14];

  // Output layout: new_state, new_output, w1n, w2n, w3n, A_ema_new (flat concat)
  float* out = (float*)d_out;
  float* ns  = out;                       // [B,N,D] target
  float* no  = out + (size_t)SND;         // [B,N,D] prediction
  float* w1o = out + 2 * (size_t)SND;
  float* w2o = w1o + SWD;
  float* w3o = w2o + SWD;
  float* aeo = w3o + SWD;                 // [B,N,N] A_ema_new

  // Workspace layout (bytes)
  char* W = (char*)d_ws;
  float* q      = (float*)(W + (size_t)0);          // queries  1MB
  float* kbuf   = (float*)(W + (size_t)(1  << 20)); // keys     1MB
  float* rawA   = (float*)(W + (size_t)(2  << 20)); // 4MB
  float* Aw     = (float*)(W + (size_t)(6  << 20)); // sinkhorn work, 4MB
  float* err1   = (float*)(W + (size_t)(10 << 20));
  float* err2   = (float*)(W + (size_t)(11 << 20));
  float* err3   = (float*)(W + (size_t)(12 << 20));
  float* noisy1 = (float*)(W + (size_t)(13 << 20));
  float* noisy2 = (float*)(W + (size_t)(14 << 20));
  float* noisy3 = (float*)(W + (size_t)(15 << 20));
  float* coef1  = (float*)(W + (size_t)(16 << 20));
  float* coef2  = (float*)(W + (size_t)(17 << 20));
  float* coef3  = (float*)(W + (size_t)(18 << 20));
  _Float16* qh  = (_Float16*)(W + (size_t)(19 << 20));
  _Float16* kh  = (_Float16*)(W + (size_t)(19 << 20) + (1 << 19));
  _Float16* Ah  = (_Float16*)(W + (size_t)(20 << 20));  // 2MB
  _Float16* OhT = (_Float16*)(W + (size_t)(22 << 20));  // 0.5MB, [B,D,N]
  float* ssq    = (float*)(W + (size_t)(23 << 20));     // [B]
  float* vsum   = ssq + B_;
  float* vsq    = vsum + B_;
  float* csum   = (float*)(W + (size_t)(23 << 20) + 4096);  // [B,N] col sums

  hipMemsetAsync(ssq, 0, 3 * B_ * sizeof(float), stream);

  // 1) forward: prediction(no), keys, queries   (reads 384MB of W)
  fwd_kernel<<<dim3(B_ * N_, 3), 128, 0, stream>>>(state, w1, w2, w3, no, kbuf, q);

  // 2) raw_A = Q K^T / sqrt(D) via WMMA
  tohalf_qk_kernel<<<SND / 256, 256, 0, stream>>>(q, kbuf, qh, kh);
  wmma_qk_kernel<<<dim3(N_ / 16, N_ / 16, B_), 32, 0, stream>>>(qh, kh, rawA);

  // 3) rms_norm over (-1,-2) + EMA -> A_ema_new (output) and Aw (work copy)
  ssq_kernel<<<dim3(64, B_), 256, 0, stream>>>(rawA, ssq);
  ema_kernel<<<SNN / 256, 256, 0, stream>>>(rawA, ssq, A_ema, step, aeo, Aw);

  // 4) sinkhorn #1, threshold, sinkhorn #2, zero rows 0..1
  for (int it = 0; it < 5; ++it) {
    sink_row_kernel<<<B_ * N_, 128, 0, stream>>>(Aw);
    hipMemsetAsync(csum, 0, B_ * N_ * sizeof(float), stream);
    colsum_kernel<<<dim3(64, B_), 256, 0, stream>>>(Aw, csum);
    coldiv_kernel<<<SNN / 256, 256, 0, stream>>>(Aw, csum);
  }
  thresh_kernel<<<SNN / 256, 256, 0, stream>>>(Aw);
  for (int it = 0; it < 5; ++it) {
    sink_row_kernel<<<B_ * N_, 128, 0, stream>>>(Aw);
    hipMemsetAsync(csum, 0, B_ * N_ * sizeof(float), stream);
    colsum_kernel<<<dim3(64, B_), 256, 0, stream>>>(Aw, csum);
    coldiv_kernel<<<SNN / 256, 256, 0, stream>>>(Aw, csum);
  }
  zero_rows_kernel<<<dim3(B_, 2), N_, 0, stream>>>(Aw);

  // 5) variance -> boredom scale; target = A @ (output + noise) via WMMA
  vstat_kernel<<<dim3(64, B_), 256, 0, stream>>>(Aw, vsum, vsq);
  convA_kernel<<<SNN / 256, 256, 0, stream>>>(Aw, Ah);
  convO_kernel<<<SND / 256, 256, 0, stream>>>(outp, vsum, vsq, OhT);
  wmma_tgt_kernel<<<dim3(D_ / 16, N_ / 16, B_), 32, 0, stream>>>(Ah, OhT, ns);
  envfix_kernel<<<B_, D_, 0, stream>>>(ns, eye, stomach);

  // 6) errors, per-row gradient factors
  err_kernel<<<SND / 256, 256, 0, stream>>>(no, ns, kbuf, q, err1, err2, err3);
  coef_kernel<<<dim3(B_ * N_, 3), 128, 0, stream>>>(state, err1, err2, err3,
                                                    eb1, eb2, eb3,
                                                    noisy1, noisy2, noisy3,
                                                    coef1, coef2, coef3);

  // 7) fused weight updates (bulk of traffic: 6 reads + 3 writes of 128MB)
  wstep_kernel<<<B_ * N_, 256, 0, stream>>>(w1, g1, coef1, noisy1, step, w1o);
  wstep_kernel<<<B_ * N_, 256, 0, stream>>>(w2, g2, coef2, noisy2, step, w2o);
  wstep_kernel<<<B_ * N_, 256, 0, stream>>>(w3, g3, coef3, noisy3, step, w3o);
}
